// GNN_15650860827313
// MI455X (gfx1250) — matrix operand order repaired and checked by reading, not compiled, not run on hardware
//
#include <hip/hip_runtime.h>
#include <hip/hip_bf16.h>

typedef __attribute__((ext_vector_type(16))) _Float16 v16h;
typedef __attribute__((ext_vector_type(8)))  float    v8f;

#define N_NODES 50000
#define N_EDGES 800000
#define N_GRAPHS 512
#define HID 64

// ---------------------------------------------------------------------------
// Dual GEMM: Omsg = H @ Wrel, Oroot = H @ Wroot   (H: [n_rows x K], W: [K x 64])
// One wave computes a 16x64 tile using v_wmma_f32_16x16x32_f16.
// Weights are pre-swizzled into LDS in exact B-fragment order so each
// fragment is a single contiguous 32B LDS read (2x ds_load_b128).
// ---------------------------------------------------------------------------
template<int K>
__global__ __launch_bounds__(256)
void gemm_dual(const float* __restrict__ H,
               const float* __restrict__ Wrel,
               const float* __restrict__ Wroot,
               float* __restrict__ Omsg,
               float* __restrict__ Oroot,
               int n_rows)
{
    constexpr int KC    = K / 32;        // K chunks of 32
    constexpr int SLOTS = KC * 4 * 32;   // (kc, ntile, lane) fragment slots
    __shared__ _Float16 sW[2][SLOTS * 16];

    const int tid = threadIdx.x;

    // Cooperative swizzled weight load: slot s holds the 16 halves lane `s&31`
    // needs for B-fragment (kc = s>>7, ntile = (s>>5)&3).
    for (int s = tid; s < SLOTS; s += 256) {
        const int lane_s = s & 31;
        const int nt     = (s >> 5) & 3;
        const int kc     = s >> 7;                    // valid for KC<=4 (K<=128)
        const int n      = nt * 16 + (lane_s & 15);
        const int kb     = (lane_s < 16) ? 0 : 8;     // B layout: lanes 0-15 K=0..7,16..23
        const int kbase  = kc * 32 + kb;
        #pragma unroll
        for (int e = 0; e < 16; ++e) {
            sW[0][s * 16 + e] = (_Float16)Wrel [(kbase + e + (e >= 8 ? 8 : 0)) * 64 + n];
            sW[1][s * 16 + e] = (_Float16)Wroot[(kbase + e + (e >= 8 ? 8 : 0)) * 64 + n];
        }
    }
    __syncthreads();

    const int wave = tid >> 5;
    const int lane = tid & 31;
    const int tile = blockIdx.x * 8 + wave;          // wave-uniform
    if (tile * 16 >= n_rows) return;                 // uniform branch: EXEC all 1s inside
    const int row0 = tile * 16;

    // A-fragment addressing (16-bit A 16x32 layout):
    //   lanes 0-15 : elems 0-7 -> K=0..7,   elems 8-15 -> K=16..23
    //   lanes 16-31: elems 0-7 -> K=8..15,  elems 8-15 -> K=24..31
    const int m   = lane & 15;
    const int kbA = (lane < 16) ? 0 : 8;
    const float* hrow = H + (size_t)(row0 + m) * K;

    v8f accR[4] = {};
    v8f accT[4] = {};

    for (int kc = 0; kc < KC; ++kc) {
        const float4* p0 = (const float4*)(hrow + kc * 32 + kbA);
        const float4* p1 = (const float4*)(hrow + kc * 32 + kbA + 16);
        const float4 f0 = p0[0], f1 = p0[1];
        const float4 f2 = p1[0], f3 = p1[1];

        v16h a;
        a[0]  = (_Float16)f0.x;  a[1]  = (_Float16)f0.y;
        a[2]  = (_Float16)f0.z;  a[3]  = (_Float16)f0.w;
        a[4]  = (_Float16)f1.x;  a[5]  = (_Float16)f1.y;
        a[6]  = (_Float16)f1.z;  a[7]  = (_Float16)f1.w;
        a[8]  = (_Float16)f2.x;  a[9]  = (_Float16)f2.y;
        a[10] = (_Float16)f2.z;  a[11] = (_Float16)f2.w;
        a[12] = (_Float16)f3.x;  a[13] = (_Float16)f3.y;
        a[14] = (_Float16)f3.z;  a[15] = (_Float16)f3.w;

        #pragma unroll
        for (int nt = 0; nt < 4; ++nt) {
            const v16h bR = *(const v16h*)&sW[0][((kc * 4 + nt) * 32 + lane) * 16];
            const v16h bT = *(const v16h*)&sW[1][((kc * 4 + nt) * 32 + lane) * 16];
            accR[nt] = __builtin_amdgcn_wmma_f32_16x16x32_f16(
                false, a, false, bR, (short)0, accR[nt], false, false);
            accT[nt] = __builtin_amdgcn_wmma_f32_16x16x32_f16(
                false, a, false, bT, (short)0, accT[nt], false, false);
        }
    }

    // C/D layout: VGPR v -> M = v + 8*(lane>=16), N = lane&15
    const int rbase = row0 + ((lane >> 4) << 3);
    const int cbase = lane & 15;
    #pragma unroll
    for (int nt = 0; nt < 4; ++nt) {
        #pragma unroll
        for (int v = 0; v < 8; ++v) {
            Omsg [(size_t)(rbase + v) * 64 + nt * 16 + cbase] = accR[nt][v];
            Oroot[(size_t)(rbase + v) * 64 + nt * 16 + cbase] = accT[nt][v];
        }
    }
}

// ---------------------------------------------------------------------------
// Edge scatter: agg[dst] += msg[src], 64 features; 16 threads/edge, float4 each
// ---------------------------------------------------------------------------
__global__ __launch_bounds__(256)
void scatter_edges(const int* __restrict__ src, const int* __restrict__ dst,
                   const float* __restrict__ msg, float* __restrict__ agg,
                   int n_edges)
{
    const long long idx = (long long)blockIdx.x * blockDim.x + threadIdx.x;
    const int e  = (int)(idx >> 4);
    if (e >= n_edges) return;
    const int c4 = (int)(idx & 15) * 4;
    const int s = src[e];
    const int d = dst[e];
    const float4 v = *(const float4*)(msg + (size_t)s * 64 + c4);
    float* p = agg + (size_t)d * 64 + c4;
    atomicAdd(p + 0, v.x);
    atomicAdd(p + 1, v.y);
    atomicAdd(p + 2, v.z);
    atomicAdd(p + 3, v.w);
}

// ---------------------------------------------------------------------------
// out = relu(agg + root + bias[f])
// ---------------------------------------------------------------------------
__global__ __launch_bounds__(256)
void combine_relu(const float* __restrict__ agg, const float* __restrict__ rootv,
                  const float* __restrict__ bias, float* __restrict__ out, int n)
{
    const int i = blockIdx.x * blockDim.x + threadIdx.x;
    if (i >= n) return;
    const float v = agg[i] + rootv[i] + bias[i & 63];
    out[i] = v > 0.0f ? v : 0.0f;
}

// ---------------------------------------------------------------------------
// Mean-pool scatter: sums[batch[node]] += h[node]; cnt[batch[node]] += 1
// ---------------------------------------------------------------------------
__global__ __launch_bounds__(256)
void pool_scatter(const float* __restrict__ h, const int* __restrict__ batch,
                  float* __restrict__ sums, float* __restrict__ cnt, int n_nodes)
{
    const long long idx = (long long)blockIdx.x * blockDim.x + threadIdx.x;
    const int node = (int)(idx >> 4);
    if (node >= n_nodes) return;
    const int c4 = (int)(idx & 15) * 4;
    const int g = batch[node];
    const float4 v = *(const float4*)(h + (size_t)node * 64 + c4);
    float* p = sums + (size_t)g * 64 + c4;
    atomicAdd(p + 0, v.x);
    atomicAdd(p + 1, v.y);
    atomicAdd(p + 2, v.z);
    atomicAdd(p + 3, v.w);
    if (c4 == 0) atomicAdd(cnt + g, 1.0f);
}

// ---------------------------------------------------------------------------
// Head: pooled = sums/max(cnt,1); out = relu(pooled@W1+b1)@W2 + b2
// ---------------------------------------------------------------------------
__global__ __launch_bounds__(256)
void head_mlp(const float* __restrict__ sums, const float* __restrict__ cnt,
              const float* __restrict__ W1, const float* __restrict__ b1,
              const float* __restrict__ W2, const float* __restrict__ b2,
              float* __restrict__ out)
{
    const int g = blockIdx.x * blockDim.x + threadIdx.x;
    if (g >= N_GRAPHS) return;
    float c = cnt[g];
    c = c > 1.0f ? c : 1.0f;
    const float inv = 1.0f / c;
    float p[64];
    #pragma unroll
    for (int i = 0; i < 64; ++i) p[i] = sums[(size_t)g * 64 + i] * inv;
    float o = b2[0];
    for (int j = 0; j < 32; ++j) {
        float t = b1[j];
        #pragma unroll
        for (int i = 0; i < 64; ++i) t += p[i] * W1[i * 32 + j];
        t = t > 0.0f ? t : 0.0f;
        o += t * W2[j];
    }
    out[g] = o;
}

// ---------------------------------------------------------------------------
extern "C" void kernel_launch(void* const* d_in, const int* in_sizes, int n_in,
                              void* d_out, int out_size, void* d_ws, size_t ws_size,
                              hipStream_t stream)
{
    const float* x      = (const float*)d_in[0];
    const int*   eidx   = (const int*)  d_in[1];   // [2][E]
    const int*   batch  = (const int*)  d_in[2];
    const float* Wrel1  = (const float*)d_in[3];
    const float* brel1  = (const float*)d_in[4];
    const float* Wroot1 = (const float*)d_in[5];
    const float* Wrel2  = (const float*)d_in[6];
    const float* brel2  = (const float*)d_in[7];
    const float* Wroot2 = (const float*)d_in[8];
    const float* Wrel3  = (const float*)d_in[9];
    const float* brel3  = (const float*)d_in[10];
    const float* Wroot3 = (const float*)d_in[11];
    const float* Wfc1   = (const float*)d_in[12];
    const float* bfc1   = (const float*)d_in[13];
    const float* Wfc2   = (const float*)d_in[14];
    const float* bfc2   = (const float*)d_in[15];
    float* out = (float*)d_out;

    const int* src = eidx;
    const int* dst = eidx + N_EDGES;

    const size_t NF = (size_t)N_NODES * HID;
    float* ws    = (float*)d_ws;
    float* msg   = ws;                //  12.8 MB
    float* rootv = msg   + NF;        //  12.8 MB
    float* agg   = rootv + NF;        //  12.8 MB
    float* hA    = agg   + NF;        //  12.8 MB
    float* hB    = hA    + NF;        //  12.8 MB
    float* sums  = hB    + NF;        //  128 KB
    float* cnt   = sums  + (size_t)N_GRAPHS * HID;

    const dim3 blk(256);
    const int tiles      = N_NODES / 16;              // 3125 exactly
    const int gemmBlocks = (tiles + 7) / 8;           // 8 waves/block
    const int scatBlocks = (N_EDGES * 16) / 256;      // 50000
    const int combBlocks = (int)(NF / 256);           // 12500
    const int poolBlocks = (N_NODES * 16) / 256;      // 3125

    // ---- Layer 1 (K=128) ----
    gemm_dual<128><<<gemmBlocks, blk, 0, stream>>>(x, Wrel1, Wroot1, msg, rootv, N_NODES);
    hipMemsetAsync(agg, 0, NF * sizeof(float), stream);
    scatter_edges<<<scatBlocks, blk, 0, stream>>>(src, dst, msg, agg, N_EDGES);
    combine_relu<<<combBlocks, blk, 0, stream>>>(agg, rootv, brel1, hA, (int)NF);

    // ---- Layer 2 (K=64) ----
    gemm_dual<64><<<gemmBlocks, blk, 0, stream>>>(hA, Wrel2, Wroot2, msg, rootv, N_NODES);
    hipMemsetAsync(agg, 0, NF * sizeof(float), stream);
    scatter_edges<<<scatBlocks, blk, 0, stream>>>(src, dst, msg, agg, N_EDGES);
    combine_relu<<<combBlocks, blk, 0, stream>>>(agg, rootv, brel2, hB, (int)NF);

    // ---- Layer 3 (K=64) ----
    gemm_dual<64><<<gemmBlocks, blk, 0, stream>>>(hB, Wrel3, Wroot3, msg, rootv, N_NODES);
    hipMemsetAsync(agg, 0, NF * sizeof(float), stream);
    scatter_edges<<<scatBlocks, blk, 0, stream>>>(src, dst, msg, agg, N_EDGES);
    combine_relu<<<combBlocks, blk, 0, stream>>>(agg, rootv, brel3, hA, (int)NF);

    // ---- Mean pool + MLP head ----
    hipMemsetAsync(sums, 0, ((size_t)N_GRAPHS * HID + N_GRAPHS) * sizeof(float), stream);
    pool_scatter<<<poolBlocks, blk, 0, stream>>>(hA, batch, sums, cnt, N_NODES);
    head_mlp<<<2, blk, 0, stream>>>(sums, cnt, Wfc1, bfc1, Wfc2, bfc2, out);
}
